// SeqBoatUnit_50285477102155
// MI455X (gfx1250) — compile-verified, matrix operand order checked by reading, not run on hardware
//
#include <hip/hip_runtime.h>

typedef __bf16 bf16;
typedef __attribute__((ext_vector_type(16))) __bf16 v16bf;
typedef __attribute__((ext_vector_type(8)))  __bf16 v8bf;
typedef __attribute__((ext_vector_type(4)))  __bf16 v4bf;
typedef __attribute__((ext_vector_type(8)))  float   v8f;

#define BB  4
#define LL  4096
#define DD  1024
#define ZZ  256
#define HH  2048
#define NE  16
#define WWIN 512
#define TOK (BB*LL)

__device__ __forceinline__ float siluf(float v)    { return v / (1.0f + __expf(-v)); }
__device__ __forceinline__ float sigmf(float v)    { return 1.0f / (1.0f + __expf(-v)); }

// ---------------------------------------------------------------------------
// WMMA helpers: 16-bit A-matrix 16x32 fragment (ISA 7.12.2 layout).
// lane 0-15  : row = lane,   K = {0..7} and {16..23}
// lane 16-31 : row = lane-16,K = {8..15} and {24..31}
// B operand is identical layout on a pre-transposed (N-major) matrix.
// ---------------------------------------------------------------------------
__device__ __forceinline__ v16bf load_frag(const bf16* __restrict__ tile, int ld, int lane) {
  int row = lane & 15;
  int kb  = (lane >> 4) << 3;            // 0 or 8
  const bf16* p = tile + (long)row * ld + kb;
  v8bf lo = *reinterpret_cast<const v8bf*>(p);
  v8bf hi = *reinterpret_cast<const v8bf*>(p + 16);
  v16bf r;
#pragma unroll
  for (int i = 0; i < 8; ++i) { r[i] = lo[i]; r[i + 8] = hi[i]; }
  return r;
}

#define WMMA_BF16(a, b, c) \
  __builtin_amdgcn_wmma_f32_16x16x32_bf16(false, (a), false, (b), (short)0, (c), false, false)

// Wave tile: 32 (M) x 64 (N); 8 WMMAs per 32-wide K step.
__device__ __forceinline__ void gemm_loop(const bf16* __restrict__ A, int lda,
                                          const bf16* __restrict__ B, int ldb,
                                          int kElems, int lane, v8f acc[2][4]) {
  for (int k = 0; k < kElems; k += 32) {
    v16bf a0 = load_frag(A + k, lda, lane);
    v16bf a1 = load_frag(A + (long)16 * lda + k, lda, lane);
    v16bf b0 = load_frag(B + k, ldb, lane);
    v16bf b1 = load_frag(B + (long)16 * ldb + k, ldb, lane);
    v16bf b2 = load_frag(B + (long)32 * ldb + k, ldb, lane);
    v16bf b3 = load_frag(B + (long)48 * ldb + k, ldb, lane);
    acc[0][0] = WMMA_BF16(a0, b0, acc[0][0]);
    acc[0][1] = WMMA_BF16(a0, b1, acc[0][1]);
    acc[0][2] = WMMA_BF16(a0, b2, acc[0][2]);
    acc[0][3] = WMMA_BF16(a0, b3, acc[0][3]);
    acc[1][0] = WMMA_BF16(a1, b0, acc[1][0]);
    acc[1][1] = WMMA_BF16(a1, b1, acc[1][1]);
    acc[1][2] = WMMA_BF16(a1, b2, acc[1][2]);
    acc[1][3] = WMMA_BF16(a1, b3, acc[1][3]);
  }
}

// C/D frag element (i,j,r): local row = i*16 + r + (lane>=16 ? 8:0), col = j*16 + (lane&15)

// ---------------------------------------------------------------------------
// 1) LayerNorm: one block (256 thr) per token row of 1024
// ---------------------------------------------------------------------------
__global__ void k_ln(const float* __restrict__ x, const float* __restrict__ g,
                     const float* __restrict__ b, float* __restrict__ xn,
                     bf16* __restrict__ xnbf) {
  __shared__ float red[16];
  int row = blockIdx.x;
  int tid = threadIdx.x;
  const float4* xr = reinterpret_cast<const float4*>(x + (long)row * DD);
  float4 v = xr[tid];
  float s  = v.x + v.y + v.z + v.w;
  float sq = v.x * v.x + v.y * v.y + v.z * v.z + v.w * v.w;
#pragma unroll
  for (int m = 16; m >= 1; m >>= 1) { s += __shfl_xor(s, m, 32); sq += __shfl_xor(sq, m, 32); }
  int wid = tid >> 5;
  if ((tid & 31) == 0) { red[wid] = s; red[8 + wid] = sq; }
  __syncthreads();
  if (tid < 32) {
    float a  = (tid < 8) ? red[tid] : 0.0f;
    float a2 = (tid < 8) ? red[8 + tid] : 0.0f;
#pragma unroll
    for (int m = 4; m >= 1; m >>= 1) { a += __shfl_xor(a, m, 32); a2 += __shfl_xor(a2, m, 32); }
    if (tid == 0) { red[0] = a; red[1] = a2; }
  }
  __syncthreads();
  float mean = red[0] * (1.0f / DD);
  float var  = red[1] * (1.0f / DD) - mean * mean;
  float rstd = rsqrtf(var + 1e-5f);
  int c0 = tid * 4;
  float4 gg = reinterpret_cast<const float4*>(g)[tid];
  float4 bb = reinterpret_cast<const float4*>(b)[tid];
  float o0 = (v.x - mean) * rstd * gg.x + bb.x;
  float o1 = (v.y - mean) * rstd * gg.y + bb.y;
  float o2 = (v.z - mean) * rstd * gg.z + bb.z;
  float o3 = (v.w - mean) * rstd * gg.w + bb.w;
  reinterpret_cast<float4*>(xn + (long)row * DD)[tid] = make_float4(o0, o1, o2, o3);
  v4bf ob; ob[0] = (bf16)o0; ob[1] = (bf16)o1; ob[2] = (bf16)o2; ob[3] = (bf16)o3;
  *reinterpret_cast<v4bf*>(xnbf + (long)row * DD + c0) = ob;
}

// ---------------------------------------------------------------------------
// 2) MultiHeadEMA as linear recurrence: one lane per (b,d), N=16 states in VGPRs
// ---------------------------------------------------------------------------
__global__ void k_ema(const float* __restrict__ xn, const float* __restrict__ dlt,
                      const float* __restrict__ alp, const float* __restrict__ bet,
                      const float* __restrict__ gam, const float* __restrict__ omg,
                      bf16* __restrict__ mxbf) {
  int d = blockIdx.x * 256 + threadIdx.x;   // channel
  int b = blockIdx.y;                       // batch
  float q[NE], pb[NE], g[NE], s[NE];
#pragma unroll
  for (int n = 0; n < NE; ++n) {
    float p = sigmf(dlt[(long)d * NE + n]);
    q[n]  = 1.0f - p * sigmf(alp[(long)d * NE + n]);
    pb[n] = p * bet[(long)d * NE + n];
    g[n]  = gam[(long)d * NE + n];
    s[n]  = 0.0f;
  }
  float om = omg[d];
  const float* xp = xn + ((long)b * LL) * DD + d;
  bf16* mp = mxbf + ((long)b * LL) * DD + d;
  for (int t = 0; t < LL; ++t) {
    float xv  = xp[(long)t * DD];
    float acc = 0.0f;
#pragma unroll
    for (int n = 0; n < NE; ++n) { s[n] = q[n] * s[n] + pb[n] * xv; acc += g[n] * s[n]; }
    mp[(long)t * DD] = (bf16)(om * xv + 0.25f * acc);   // N^-0.5 = 0.25
  }
}

// ---------------------------------------------------------------------------
// 3) Weight transpose + bf16 convert (one-time prep; weights stay L2-resident)
//    dst[n*K + k] = src[k*Nn + n]
// ---------------------------------------------------------------------------
__global__ void k_tr(const float* __restrict__ src, bf16* __restrict__ dst, int K, int Nn) {
  long idx = (long)blockIdx.x * 256 + threadIdx.x;
  long tot = (long)K * Nn;
  if (idx >= tot) return;
  int k = (int)(idx / Nn), n = (int)(idx % Nn);
  dst[(long)n * K + k] = (bf16)src[idx];
}

// ---------------------------------------------------------------------------
// 4) base = silu(mx @ w_mx + b) ; split into q/k (scaled), r, vT (transposed)
// ---------------------------------------------------------------------------
__global__ void k_gemm_base(const bf16* __restrict__ A, const bf16* __restrict__ BT,
                            const float* __restrict__ bmx,
                            const float* __restrict__ qkg, const float* __restrict__ qkb,
                            bf16* __restrict__ qbf, bf16* __restrict__ kbf,
                            bf16* __restrict__ rbf, bf16* __restrict__ vT) {
  int tid = threadIdx.x, lane = tid & 31, wave = tid >> 5;
  int wm = wave >> 1, wn = wave & 1;
  int m0 = blockIdx.x * 128 + wm * 32;
  int n0 = blockIdx.y * 128 + wn * 64;
  v8f acc[2][4] = {};
  gemm_loop(A + (long)m0 * DD, DD, BT + (long)n0 * DD, DD, DD, lane, acc);
  int lhi8 = (lane >> 4) << 3, lc = lane & 15;
#pragma unroll
  for (int i = 0; i < 2; ++i)
#pragma unroll
    for (int j = 0; j < 4; ++j)
#pragma unroll
      for (int r = 0; r < 8; ++r) {
        int row = m0 + i * 16 + r + lhi8;
        int col = n0 + j * 16 + lc;
        float v = siluf(acc[i][j][r] + bmx[col]);
        if (col < ZZ) {
          float qv = (v * qkg[col] + qkb[col]) * 0.0625f;   // Z^-0.5
          float kv =  v * qkg[ZZ + col] + qkb[ZZ + col];
          qbf[(long)row * ZZ + col] = (bf16)qv;
          kbf[(long)row * ZZ + col] = (bf16)kv;
        } else if (col < ZZ + HH) {
          rbf[(long)row * HH + (col - ZZ)] = (bf16)v;
        } else {
          int h = col - (ZZ + HH);
          int bb = row >> 12, t = row & (LL - 1);
          vT[((long)bb * HH + h) * LL + t] = (bf16)v;       // h-major for PV B operand
        }
      }
}

// ---------------------------------------------------------------------------
// 5) scores S = relu2(mask(q @ k^T + relpos)) per (b, window), K = Z = 256
// ---------------------------------------------------------------------------
__global__ void k_sim(const bf16* __restrict__ qbf, const bf16* __restrict__ kbf,
                      const float* __restrict__ relpos, bf16* __restrict__ S) {
  int tid = threadIdx.x, lane = tid & 31, wave = tid >> 5;
  int wm = wave >> 1, wn = wave & 1;
  int bw = blockIdx.z, b = bw >> 3, w = bw & 7;
  int m0 = blockIdx.x * 128 + wm * 32;   // query local [0,512)
  int n0 = blockIdx.y * 128 + wn * 64;   // key   local [0,1024)
  v8f acc[2][4] = {};
  bool compute = !(w == 0 && (int)blockIdx.y < 4);   // padded keys: fully masked, skip
  if (compute) {
    const bf16* A  = qbf + ((long)(b * LL + w * WWIN + m0)) * ZZ;
    const bf16* Bp = kbf + ((long)(b * LL + (w - 1) * WWIN + n0)) * ZZ;
    gemm_loop(A, ZZ, Bp, ZZ, ZZ, lane, acc);
  }
  int lhi8 = (lane >> 4) << 3, lc = lane & 15;
  bf16* Sout = S + (long)bw * WWIN * (2 * WWIN);
#pragma unroll
  for (int i = 0; i < 2; ++i)
#pragma unroll
    for (int j = 0; j < 4; ++j)
#pragma unroll
      for (int r = 0; r < 8; ++r) {
        int iq = m0 + i * 16 + r + lhi8;
        int jk = n0 + j * 16 + lc;
        // valid: tk <= tq <= tk+W and not padded  ==>  iq <= jk <= iq+W, (w>0 || jk>=W)
        bool valid = (jk >= iq) && (jk <= iq + WWIN) && (w > 0 || jk >= WWIN);
        float sv = 0.0f;
        if (valid) {
          float t = acc[i][j][r] + relpos[1535 + jk - iq];  // MAXP-1 = 1535
          t = fmaxf(t, 0.0f);
          sv = t * t;
        }
        Sout[(long)iq * (2 * WWIN) + jk] = (bf16)sv;
      }
}

// ---------------------------------------------------------------------------
// 6) attn_out = S @ v_la ; fused epilogue ar = attn_out * r  (bf16)
// ---------------------------------------------------------------------------
__global__ void k_pv(const bf16* __restrict__ S, const bf16* __restrict__ vT,
                     const bf16* __restrict__ rbf, bf16* __restrict__ ar) {
  int tid = threadIdx.x, lane = tid & 31, wave = tid >> 5;
  int wm = wave >> 1, wn = wave & 1;
  int bw = blockIdx.z, b = bw >> 3, w = bw & 7;
  int m0 = blockIdx.x * 128 + wm * 32;   // query local
  int n0 = blockIdx.y * 128 + wn * 64;   // head channel
  int kstart = (w == 0) ? WWIN : 0;      // skip padded (all-zero) K range
  const bf16* A  = S + (long)bw * WWIN * (2 * WWIN) + (long)m0 * (2 * WWIN) + kstart;
  const bf16* Bp = vT + ((long)b * HH + n0) * LL + ((w - 1) * WWIN + kstart);
  v8f acc[2][4] = {};
  gemm_loop(A, 2 * WWIN, Bp, LL, 2 * WWIN - kstart, lane, acc);
  int lhi8 = (lane >> 4) << 3, lc = lane & 15;
#pragma unroll
  for (int i = 0; i < 2; ++i)
#pragma unroll
    for (int j = 0; j < 4; ++j)
#pragma unroll
      for (int r = 0; r < 8; ++r) {
        int iq = m0 + i * 16 + r + lhi8;
        int h  = n0 + j * 16 + lc;
        long row = (long)b * LL + w * WWIN + iq;
        float rv = (float)rbf[row * HH + h];
        ar[row * HH + h] = (bf16)(acc[i][j][r] * rv);
      }
}

// ---------------------------------------------------------------------------
// 7) out = x + silu(xn@w_hx + ar@w_h + b_hx + b_h)  (two K-loops, one accum)
// ---------------------------------------------------------------------------
__global__ void k_out(const bf16* __restrict__ xnbf, const bf16* __restrict__ whxT,
                      const bf16* __restrict__ arbf, const bf16* __restrict__ whT,
                      const float* __restrict__ bhx, const float* __restrict__ bh,
                      const float* __restrict__ x, float* __restrict__ out) {
  int tid = threadIdx.x, lane = tid & 31, wave = tid >> 5;
  int wm = wave >> 1, wn = wave & 1;
  int m0 = blockIdx.x * 128 + wm * 32;
  int n0 = blockIdx.y * 128 + wn * 64;
  v8f acc[2][4] = {};
  gemm_loop(xnbf + (long)m0 * DD, DD, whxT + (long)n0 * DD, DD, DD, lane, acc);
  gemm_loop(arbf + (long)m0 * HH, HH, whT + (long)n0 * HH, HH, HH, lane, acc);
  int lhi8 = (lane >> 4) << 3, lc = lane & 15;
#pragma unroll
  for (int i = 0; i < 2; ++i)
#pragma unroll
    for (int j = 0; j < 4; ++j)
#pragma unroll
      for (int r = 0; r < 8; ++r) {
        int row = m0 + i * 16 + r + lhi8;
        int col = n0 + j * 16 + lc;
        float v = acc[i][j][r] + bhx[col] + bh[col];
        out[(long)row * DD + col] = x[(long)row * DD + col] + siluf(v);
      }
}

// ---------------------------------------------------------------------------
extern "C" void kernel_launch(void* const* d_in, const int* in_sizes, int n_in,
                              void* d_out, int out_size, void* d_ws, size_t ws_size,
                              hipStream_t stream) {
  (void)in_sizes; (void)n_in; (void)out_size; (void)ws_size;
  const float* x      = (const float*)d_in[0];
  const float* ln_g   = (const float*)d_in[1];
  const float* ln_b   = (const float*)d_in[2];
  const float* e_del  = (const float*)d_in[3];
  const float* e_alp  = (const float*)d_in[4];
  const float* e_bet  = (const float*)d_in[5];
  const float* e_gam  = (const float*)d_in[6];
  const float* e_om   = (const float*)d_in[7];
  const float* qkg    = (const float*)d_in[8];
  const float* qkb    = (const float*)d_in[9];
  const float* w_mx   = (const float*)d_in[10];
  const float* b_mx   = (const float*)d_in[11];
  const float* w_hx   = (const float*)d_in[12];
  const float* b_hx   = (const float*)d_in[13];
  const float* w_h    = (const float*)d_in[14];
  const float* b_h    = (const float*)d_in[15];
  const float* relpos = (const float*)d_in[16];
  float* out = (float*)d_out;

  char* ws = (char*)d_ws;
  size_t off = 0;
  auto carve = [&](size_t bytes) -> void* {
    void* p = ws + off;
    off = (off + bytes + 255) & ~(size_t)255;
    return p;
  };
  float* xn  = (float*)carve((size_t)TOK * DD * 4);
  bf16* xnbf = (bf16*)carve((size_t)TOK * DD * 2);
  bf16* mxbf = (bf16*)carve((size_t)TOK * DD * 2);
  bf16* qbf  = (bf16*)carve((size_t)TOK * ZZ * 2);
  bf16* kbf  = (bf16*)carve((size_t)TOK * ZZ * 2);
  bf16* rbf  = (bf16*)carve((size_t)TOK * HH * 2);
  bf16* vT   = (bf16*)carve((size_t)BB * HH * LL * 2);
  bf16* S    = (bf16*)carve((size_t)BB * 8 * WWIN * 2 * WWIN * 2);
  bf16* ar   = (bf16*)carve((size_t)TOK * HH * 2);
  bf16* wmxT = (bf16*)carve((size_t)(ZZ + 2 * HH) * DD * 2);
  bf16* whxT = (bf16*)carve((size_t)DD * DD * 2);
  bf16* whT  = (bf16*)carve((size_t)DD * HH * 2);

  // 1) layernorm (also emits bf16 copy for the hx GEMM)
  k_ln<<<dim3(TOK), dim3(256), 0, stream>>>(x, ln_g, ln_b, xn, xnbf);

  // 2) EMA recurrence -> mx (bf16, GEMM A operand)
  k_ema<<<dim3(DD / 256, BB), dim3(256), 0, stream>>>(xn, e_del, e_alp, e_bet, e_gam, e_om, mxbf);

  // 3) weight prep (transpose + bf16); stays L2-resident afterwards
  {
    long t1 = (long)DD * (ZZ + 2 * HH);
    k_tr<<<dim3((unsigned)((t1 + 255) / 256)), dim3(256), 0, stream>>>(w_mx, wmxT, DD, ZZ + 2 * HH);
    long t2 = (long)DD * DD;
    k_tr<<<dim3((unsigned)((t2 + 255) / 256)), dim3(256), 0, stream>>>(w_hx, whxT, DD, DD);
    long t3 = (long)HH * DD;
    k_tr<<<dim3((unsigned)((t3 + 255) / 256)), dim3(256), 0, stream>>>(w_h, whT, HH, DD);
  }

  // 4) base GEMM + silu + q/k/r/vT split   (16384 x 4352 x 1024)
  k_gemm_base<<<dim3(TOK / 128, (ZZ + 2 * HH) / 128), dim3(256), 0, stream>>>(
      mxbf, wmxT, b_mx, qkg, qkb, qbf, kbf, rbf, vT);

  // 5) windowed scores (512 x 1024 per (b,w), K=256) with relu2 + mask + relpos
  k_sim<<<dim3(WWIN / 128, (2 * WWIN) / 128, BB * 8), dim3(256), 0, stream>>>(qbf, kbf, relpos, S);

  // 6) PV GEMM (512 x 2048 x 1024 per (b,w)) fused with * r
  k_pv<<<dim3(WWIN / 128, HH / 128, BB * 8), dim3(256), 0, stream>>>(S, vT, rbf, ar);

  // 7) fused output: xn@w_hx + ar@w_h + biases -> silu -> +x
  k_out<<<dim3(TOK / 128, DD / 128), dim3(256), 0, stream>>>(xnbf, whxT, ar, whT, b_hx, b_h, x, out);
}
